// LightningIndexer_17755394801796
// MI455X (gfx1250) — matrix operand order repaired and checked by reading, not compiled, not run on hardware
//
#include <hip/hip_runtime.h>
#include <hip/hip_bf16.h>
#include <stdint.h>

// ---- problem constants -------------------------------------------------
#define SEQ   4096
#define DM    2048
#define NH    4
#define DH    64
#define NB    2
#define ROWS  (NB * SEQ)        // 8192 flattened (b, s) rows
#define QKC   (NH * DH + DH)    // 320 = 256 q features + 64 k features

// ---- types -------------------------------------------------------------
typedef __attribute__((ext_vector_type(16))) __bf16       v16bf;
typedef __attribute__((ext_vector_type(8)))  float        v8f;
typedef __attribute__((ext_vector_type(4)))  unsigned int v4u;

union Frag { v16bf bf; v4u u[2]; };

__device__ inline v8f zero8() {
  v8f z;
#pragma unroll
  for (int j = 0; j < 8; ++j) z[j] = 0.0f;
  return z;
}

__device__ inline uint16_t f2bf(float x) {
  unsigned u = __float_as_uint(x);
  return (uint16_t)((u + 0x7FFFu + ((u >> 16) & 1u)) >> 16);
}

// A-matrix fragment (16x32, M x K, 16-bit): lane L holds row r0 + (L&15);
// dwords 0-3 = K [k + (L>>4)*8, +8), dwords 4-7 = K [k + 16 + (L>>4)*8, +8)
__device__ inline v16bf load_a_frag(const uint16_t* __restrict__ base, int stride,
                                    int row, int k, int lane) {
  const uint16_t* p = base + (size_t)(row + (lane & 15)) * stride
                           + (k + ((lane >> 4) << 3));
  Frag f;
  f.u[0] = *(const v4u*)(p);
  f.u[1] = *(const v4u*)(p + 16);
  return f.bf;
}

// B-matrix fragment (32x16, K x N, 16-bit, column-major source = row-major
// [n, k] memory): lane L holds col n0 + (L&15);
// dwords 0-3 = K [k + (L>>4)*8, +8), dwords 4-7 = K [k + 16 + (L>>4)*8, +8)
__device__ inline v16bf load_b_frag(const uint16_t* __restrict__ base, int stride,
                                    int col, int k, int lane) {
  const uint16_t* p = base + (size_t)(col + (lane & 15)) * stride
                           + (k + ((lane >> 4) << 3));
  Frag f;
  f.u[0] = *(const v4u*)(p);
  f.u[1] = *(const v4u*)(p + 16);
  return f.bf;
}

// ---- fp32 -> bf16 conversion -------------------------------------------
__global__ void cvt_f32_bf16(const float* __restrict__ in,
                             uint16_t* __restrict__ out, int n) {
  int i = blockIdx.x * blockDim.x + threadIdx.x;
  if (i < n) out[i] = f2bf(in[i]);
}

// ---- stage 1: qk[row, 0:320] = h_bf16[row, :] @ W_bf16[c, :]^T ----------
// wave tile 64x64, workgroup = 8 waves, grid covers (ROWS/64) x (QKC/64)
__global__ __launch_bounds__(256) void proj_wmma(
    const uint16_t* __restrict__ hbf,   // [ROWS, DM]
    const uint16_t* __restrict__ Wbf,   // [QKC, DM]
    uint16_t* __restrict__ qk) {        // [ROWS, QKC]
  const int lane = threadIdx.x & 31;
  const int wid  = blockIdx.x * 8 + (threadIdx.x >> 5);
  const int nCol = QKC / 64;            // 5
  const int r0   = (wid / nCol) * 64;
  const int c0   = (wid % nCol) * 64;

  v8f acc[4][4];
#pragma unroll
  for (int m = 0; m < 4; ++m)
#pragma unroll
    for (int n = 0; n < 4; ++n) acc[m][n] = zero8();

  for (int k = 0; k < DM; k += 32) {
    v16bf a[4], b[4];
#pragma unroll
    for (int m = 0; m < 4; ++m) a[m] = load_a_frag(hbf, DM, r0 + 16 * m, k, lane);
#pragma unroll
    for (int n = 0; n < 4; ++n) b[n] = load_b_frag(Wbf, DM, c0 + 16 * n, k, lane);
#pragma unroll
    for (int m = 0; m < 4; ++m)
#pragma unroll
      for (int n = 0; n < 4; ++n)
        acc[m][n] = __builtin_amdgcn_wmma_f32_16x16x32_bf16(
            false, a[m], false, b[n], (short)0, acc[m][n], false, false);
  }

  const int rl = (lane >> 4) << 3;      // row offset within tile: 0 or 8
  const int cl = lane & 15;             // col within tile
#pragma unroll
  for (int m = 0; m < 4; ++m)
#pragma unroll
    for (int n = 0; n < 4; ++n)
#pragma unroll
      for (int j = 0; j < 8; ++j) {
        const int row = r0 + 16 * m + rl + j;
        const int col = c0 + 16 * n + cl;
        qk[(size_t)row * QKC + col] = f2bf(acc[m][n][j]);
      }
}

// ---- stage 2: I[b,t,s] = sum_h w[h] * relu(q_h[t] . k[s]) ---------------
// wave tile 64 (t) x 32 (s); per head: K=64 = 2 WMMA k-steps, relu+fma epi.
__global__ __launch_bounds__(256) void indexer_wmma(
    const uint16_t* __restrict__ qk,    // [NB, SEQ, QKC]
    const float* __restrict__ w,        // [NH]
    float* __restrict__ out) {          // [NB, SEQ, SEQ]
  const int lane = threadIdx.x & 31;
  const int wid  = blockIdx.x * 8 + (threadIdx.x >> 5);
  const int tilesS   = SEQ / 32;        // 128
  const int perBatch = (SEQ / 64) * tilesS; // 8192
  const int b  = wid / perBatch;
  const int r  = wid % perBatch;
  const int t0 = (r / tilesS) * 64;
  const int s0 = (r % tilesS) * 32;

  const uint16_t* qb = qk + (size_t)b * SEQ * QKC;       // q features 0..255
  const uint16_t* kb = qb + NH * DH;                     // k features 256..319

  v8f fin[4][2];
#pragma unroll
  for (int m = 0; m < 4; ++m)
#pragma unroll
    for (int n = 0; n < 2; ++n) fin[m][n] = zero8();

#pragma unroll
  for (int h = 0; h < NH; ++h) {
    const float wh = w[h];
    v8f c[4][2];
#pragma unroll
    for (int m = 0; m < 4; ++m)
#pragma unroll
      for (int n = 0; n < 2; ++n) c[m][n] = zero8();

#pragma unroll
    for (int k = 0; k < DH; k += 32) {
      v16bf a[4], bb[2];
#pragma unroll
      for (int m = 0; m < 4; ++m)
        a[m] = load_a_frag(qb, QKC, t0 + 16 * m, h * DH + k, lane);
#pragma unroll
      for (int n = 0; n < 2; ++n)
        bb[n] = load_b_frag(kb, QKC, s0 + 16 * n, k, lane);
#pragma unroll
      for (int m = 0; m < 4; ++m)
#pragma unroll
        for (int n = 0; n < 2; ++n)
          c[m][n] = __builtin_amdgcn_wmma_f32_16x16x32_bf16(
              false, a[m], false, bb[n], (short)0, c[m][n], false, false);
    }
    // relu + head-weighted accumulate
#pragma unroll
    for (int m = 0; m < 4; ++m)
#pragma unroll
      for (int n = 0; n < 2; ++n)
#pragma unroll
        for (int j = 0; j < 8; ++j)
          fin[m][n][j] = fmaf(wh, fmaxf(c[m][n][j], 0.0f), fin[m][n][j]);
  }

  float* ob = out + (size_t)b * SEQ * SEQ;
  const int rl = (lane >> 4) << 3;
  const int cl = lane & 15;
#pragma unroll
  for (int m = 0; m < 4; ++m)
#pragma unroll
    for (int n = 0; n < 2; ++n)
#pragma unroll
      for (int j = 0; j < 8; ++j)
        ob[(size_t)(t0 + 16 * m + rl + j) * SEQ + (s0 + 16 * n + cl)] =
            fin[m][n][j];
}

// ---- launcher -----------------------------------------------------------
extern "C" void kernel_launch(void* const* d_in, const int* in_sizes, int n_in,
                              void* d_out, int out_size, void* d_ws, size_t ws_size,
                              hipStream_t stream) {
  const float* h  = (const float*)d_in[0];   // [NB, SEQ, DM]
  const float* Wq = (const float*)d_in[1];   // [256, DM]
  const float* Wk = (const float*)d_in[2];   // [64, DM]
  const float* w  = (const float*)d_in[3];   // [NH]
  float* out = (float*)d_out;

  // workspace layout (bf16 = uint16_t), 256B aligned slabs
  uint8_t* ws = (uint8_t*)d_ws;
  size_t offW  = 0;
  size_t szW   = (size_t)QKC * DM * sizeof(uint16_t);           // 1.25 MB
  size_t offH  = (offW + szW + 255) & ~(size_t)255;
  size_t szH   = (size_t)ROWS * DM * sizeof(uint16_t);          // 32 MB
  size_t offQK = (offH + szH + 255) & ~(size_t)255;

  uint16_t* Wbf  = (uint16_t*)(ws + offW);
  uint16_t* hbf  = (uint16_t*)(ws + offH);
  uint16_t* qkbf = (uint16_t*)(ws + offQK);

  // 1-2: fp32 -> bf16 conversions (W = [Wq ; Wk] stacked, rows x DM)
  {
    int n = 256 * DM;
    cvt_f32_bf16<<<(n + 255) / 256, 256, 0, stream>>>(Wq, Wbf, n);
  }
  {
    int n = 64 * DM;
    cvt_f32_bf16<<<(n + 255) / 256, 256, 0, stream>>>(Wk, Wbf + 256 * DM, n);
  }
  {
    int n = ROWS * DM;
    cvt_f32_bf16<<<(n + 255) / 256, 256, 0, stream>>>(h, hbf, n);
  }

  // 3: projection GEMM -> qk bf16 [ROWS, QKC]
  {
    int waves  = (ROWS / 64) * (QKC / 64);  // 128 * 5 = 640
    int blocks = waves / 8;                  // 80
    proj_wmma<<<blocks, 256, 0, stream>>>(hbf, Wbf, qkbf);
  }

  // 4: indexer GEMM + relu + head-weighted sum -> out fp32 [NB, SEQ, SEQ]
  {
    int waves  = NB * (SEQ / 64) * (SEQ / 32); // 2 * 64 * 128 = 16384
    int blocks = waves / 8;                     // 2048
    indexer_wmma<<<blocks, 256, 0, stream>>>(qkbf, w, out);
  }
  (void)in_sizes; (void)n_in; (void)out_size; (void)ws_size;
}
